// HNHNModel_19069654794244
// MI455X (gfx1250) — matrix operand order repaired and checked by reading, not compiled, not run on hardware
//
#include <hip/hip_runtime.h>
#include <hip/hip_bf16.h>
#include <cstddef>

typedef __attribute__((ext_vector_type(16))) _Float16 v16h;
typedef __attribute__((ext_vector_type(8)))  _Float16 v8h;
typedef __attribute__((ext_vector_type(8)))  float    v8f;

#define HDIM   64
#define DIN    128
#define KSPLIT 8

// ---------------- Stage 0: single pass over f32 B ----------------
// Builds BbitE (row-major bit pack along E), BbitN (bit pack along N, via atomicOr),
// node_deg (block reduce), edge_card (atomics on nonzeros). B f32 is read exactly once.
__global__ void k_build(const float* __restrict__ B, unsigned* __restrict__ BbitE,
                        unsigned* __restrict__ BbitN, float* __restrict__ edge_card,
                        float* __restrict__ node_deg, int E, int N) {
  __shared__ float red[256];
  const int row = blockIdx.x;
  const int wordsE = E >> 5;
  const int wordsN = N >> 5;
  const unsigned rbit = 1u << (row & 31);
  const int rword = row >> 5;
  float s = 0.f;
  for (int wi = threadIdx.x; wi < wordsE; wi += 256) {
    const int ebase = wi << 5;
    const float4* p = (const float4*)(B + (size_t)row * E + ebase);
    unsigned wd = 0u;
#pragma unroll
    for (int q = 0; q < 8; q++) {
      float4 v4 = p[q];
      float fv[4] = {v4.x, v4.y, v4.z, v4.w};
#pragma unroll
      for (int j = 0; j < 4; j++) {
        if (fv[j] != 0.f) {
          int e = ebase + q * 4 + j;
          wd |= 1u << (q * 4 + j);
          s += fv[j];
          atomicAdd(&edge_card[e], fv[j]);
          atomicOr(&BbitN[(size_t)e * wordsN + rword], rbit);
        }
      }
    }
    BbitE[(size_t)row * wordsE + wi] = wd;
  }
  red[threadIdx.x] = s;
  __syncthreads();
  for (int off = 128; off > 0; off >>= 1) {
    if (threadIdx.x < off) red[threadIdx.x] += red[threadIdx.x + off];
    __syncthreads();
  }
  if (threadIdx.x == 0) node_deg[row] = red[0];
}

// e_alpha = card^-1.5, v_beta = deg^-0.5 (1.0 when degree is zero)
__global__ void k_pows(const float* __restrict__ ec, const float* __restrict__ nd,
                       float* __restrict__ e_alpha, float* __restrict__ v_beta,
                       int E, int N) {
  int i = blockIdx.x * 256 + threadIdx.x;
  if (i < E) {
    float c = ec[i];
    e_alpha[i] = (c > 0.f) ? (rsqrtf(c) / c) : 1.f;
  }
  if (i < N) {
    float d = nd[i];
    v_beta[i] = (d > 0.f) ? rsqrtf(d) : 1.f;
  }
}

// Denominators from the 16MB bit matrix: alpha_denom per row (reduce, stored inverted);
// beta_denom per column via atomics on set bits only.
__global__ void k_denom2(const unsigned* __restrict__ BbitE, const float* __restrict__ e_alpha,
                         const float* __restrict__ v_beta, float* __restrict__ beta_denom,
                         float* __restrict__ inv_alpha_denom, int E) {
  __shared__ float red[256];
  const int row = blockIdx.x;
  const int wordsE = E >> 5;
  const float vb = v_beta[row];
  float s = 0.f;
  for (int wi = threadIdx.x; wi < wordsE; wi += 256) {
    unsigned wd = BbitE[(size_t)row * wordsE + wi];
    while (wd) {
      int b = __builtin_ctz(wd);
      int e = (wi << 5) + b;
      s += e_alpha[e];
      atomicAdd(&beta_denom[e], vb);
      wd &= wd - 1;
    }
  }
  red[threadIdx.x] = s;
  __syncthreads();
  for (int off = 128; off > 0; off >>= 1) {
    if (threadIdx.x < off) red[threadIdx.x] += red[threadIdx.x + off];
    __syncthreads();
  }
  if (threadIdx.x == 0) {
    float ad = red[0];
    inv_alpha_denom[row] = 1.f / ((ad == 0.f) ? 1.f : ad);
  }
}

__global__ void k_invfix(const float* __restrict__ bd, float* __restrict__ inv_bd, int E) {
  int i = blockIdx.x * 256 + threadIdx.x;
  if (i < E) {
    float v = bd[i];
    inv_bd[i] = 1.f / ((v == 0.f) ? 1.f : v);
  }
}

// ---------------- Small GEMM: Ut[h][r] = scale[r] * (X @ W)[r][h], f16 transposed output ----
__global__ void k_sgemm(const float* __restrict__ X, const float* __restrict__ W,
                        const float* __restrict__ scale, _Float16* __restrict__ Ut,
                        int R, int Kd) {
  extern __shared__ float xs[];                 // 32 * Kd floats
  const int t = threadIdx.x;
  const int rb = blockIdx.x * 32;
  for (int i = t; i < 32 * Kd; i += 256)
    xs[i] = X[(size_t)(rb + i / Kd) * Kd + (i % Kd)];
  __syncthreads();

  const int h = t & 63;
  const int rg = t >> 6;
  float acc[8];
#pragma unroll
  for (int j = 0; j < 8; j++) acc[j] = 0.f;

  for (int k = 0; k < Kd; k++) {
    float w = W[k * HDIM + h];
#pragma unroll
    for (int j = 0; j < 8; j++) acc[j] += xs[(rg * 8 + j) * Kd + k] * w;
  }

  v8h o;
#pragma unroll
  for (int j = 0; j < 8; j++) o[j] = (_Float16)(acc[j] * scale[rb + rg * 8 + j]);
  *(v8h*)(Ut + (size_t)h * R + rb + rg * 8) = o;
}

// ---------------- Big WMMA: acc(M x 64) += S(M x Kdim) @ U(Kdim x 64) ----------------
// S rows come from a bit-packed matrix (Bbit, rows of Kdim bits): one u64 load per lane
// covers a 64-wide K step; expansion to f16 0/1 is pure VALU. Ut (f16, 64 x Kdim) slab
// staged in LDS per block. Each wave: one 16x64 tile; grid.y splits K; atomicAdd partials.
__global__ __launch_bounds__(256) void k_wmma(
    const unsigned* __restrict__ Bbit, const _Float16* __restrict__ Ut,
    float* __restrict__ acc, int Kdim, int Kchunk) {
  __shared__ _Float16 uts[64 * 64];             // 8KB: Ut[:, kb:kb+64]
  const int lane   = threadIdx.x & 31;
  const int wave   = threadIdx.x >> 5;
  const int laneM  = lane & 15;
  const int laneHi = lane >> 4;
  const int m0 = blockIdx.x * 128 + wave * 16;
  const int kb0 = blockIdx.y * Kchunk;
  const int words = Kdim >> 5;

  v8f c[4] = {};

  const int sh    = threadIdx.x >> 2;           // 0..63 Ut row to stage
  const int spart = threadIdx.x & 3;            // 16-half chunk within 64
  const int kgA = laneHi * 16;
  const int kgB = laneHi * 16;
  const unsigned* brow = Bbit + (size_t)(m0 + laneM) * words;
  const _Float16 h1 = (_Float16)1.0f;
  const _Float16 h0 = (_Float16)0.0f;

  for (int kb = kb0; kb < kb0 + Kchunk; kb += 64) {
    // stage Ut[:, kb:kb+64] into LDS (each thread two 16B chunks, coalesced)
    *(uint4*)(uts + sh * 64 + spart * 16) =
        *(const uint4*)(Ut + (size_t)sh * Kdim + kb + spart * 16);
    *(uint4*)(uts + sh * 64 + spart * 16 + 8) =
        *(const uint4*)(Ut + (size_t)sh * Kdim + kb + spart * 16 + 8);
    __syncthreads();

    // one u64 of incidence bits covers this lane's K 0..63
    uint2 wp = *(const uint2*)(brow + (kb >> 5));
#pragma unroll
    for (int hstep = 0; hstep < 2; hstep++) {
      unsigned wv = hstep ? wp.y : wp.x;
      // A fragment: contiguous K (lanes 0-15: K 0..15; lanes 16-31: K 16..31)
      v16h a;
#pragma unroll
      for (int i = 0; i < 8; i++)
        a[i] = ((wv >> (kgA + i)) & 1u) ? h1 : h0;
#pragma unroll
      for (int i = 0; i < 8; i++)
        a[8 + i] = ((wv >> (kgA + 8 + i)) & 1u) ? h1 : h0;

#pragma unroll
      for (int t = 0; t < 4; t++) {
        v16h b = *(const v16h*)(uts + (t * 16 + laneM) * 64 + hstep * 32 + kgB);
        c[t] = __builtin_amdgcn_wmma_f32_16x16x32_f16(
            false, a, false, b, (short)0, c[t], false, false);
      }
    }
    __syncthreads();
  }

#pragma unroll
  for (int t = 0; t < 4; t++) {
#pragma unroll
    for (int j = 0; j < 8; j++) {
      int row = m0 + laneHi * 8 + j;
      int col = t * 16 + laneM;
      atomicAdd(&acc[(size_t)row * HDIM + col], c[t][j]);
    }
  }
}

// ---------------- Epilogue: buf = relu(buf * inv_denom[row] + bias[h]) ----------------
__global__ void k_epi(float* __restrict__ buf, const float* __restrict__ inv_denom,
                      const float* __restrict__ bias, int R) {
  int i = blockIdx.x * 256 + threadIdx.x;
  if (i < R * HDIM) {
    int r = i >> 6, h = i & 63;
    float v = buf[i] * inv_denom[r] + bias[h];
    buf[i] = v > 0.f ? v : 0.f;
  }
}

// ---------------- Max pool over rows (post-ReLU values >= 0 -> uint-bit atomicMax) ----
__global__ void k_pool(const float* __restrict__ x, unsigned int* __restrict__ pooled) {
  __shared__ float red[256];
  const int h = threadIdx.x & 63;
  const int rg = threadIdx.x >> 6;
  const int base = blockIdx.x * 256;
  float m = 0.f;
  for (int rr = rg; rr < 256; rr += 4) {
    float v = x[(size_t)(base + rr) * HDIM + h];
    m = v > m ? v : m;
  }
  red[threadIdx.x] = m;
  __syncthreads();
  if (rg == 0) {
    m = fmaxf(fmaxf(red[h], red[64 + h]), fmaxf(red[128 + h], red[192 + h]));
    atomicMax(&pooled[h], __float_as_uint(m));
  }
}

__global__ void k_final(const unsigned int* __restrict__ pooled,
                        const float* __restrict__ Wout, const float* __restrict__ bout,
                        float* __restrict__ out) {
  __shared__ float red[64];
  int h = threadIdx.x;
  red[h] = __uint_as_float(pooled[h]) * Wout[h];
  __syncthreads();
  for (int off = 32; off > 0; off >>= 1) {
    if (h < off) red[h] += red[h + off];
    __syncthreads();
  }
  if (h == 0) out[0] = red[0] + bout[0];
}

// ---------------- Host ----------------
extern "C" void kernel_launch(void* const* d_in, const int* in_sizes, int n_in,
                              void* d_out, int out_size, void* d_ws, size_t ws_size,
                              hipStream_t stream) {
  (void)n_in; (void)out_size; (void)ws_size;
  const float* x0   = (const float*)d_in[0];
  const float* B    = (const float*)d_in[1];
  const float* W01[2] = {(const float*)d_in[2], (const float*)d_in[6]};
  const float* b01[2] = {(const float*)d_in[3], (const float*)d_in[7]};
  const float* W10[2] = {(const float*)d_in[4], (const float*)d_in[8]};
  const float* b10[2] = {(const float*)d_in[5], (const float*)d_in[9]};
  const float* Wout = (const float*)d_in[10];
  const float* bout = (const float*)d_in[11];

  const int N = in_sizes[0] / DIN;      // 8192
  const int E = in_sizes[1] / N;        // 16384

  // workspace carve (256B aligned)
  char* w = (char*)d_ws;
  auto carve = [&](size_t bytes) -> char* {
    char* p = w;
    w += (bytes + 255) & ~(size_t)255;
    return p;
  };
  float* edge_card  = (float*)carve((size_t)E * 4);
  float* node_deg   = (float*)carve((size_t)N * 4);
  float* e_alpha    = (float*)carve((size_t)E * 4);
  float* v_beta     = (float*)carve((size_t)N * 4);
  float* beta_denom = (float*)carve((size_t)E * 4);
  float* inv_alpha  = (float*)carve((size_t)N * 4);
  float* inv_beta   = (float*)carve((size_t)E * 4);
  unsigned* BbitE   = (unsigned*)carve((size_t)N * (E >> 5) * 4);   // 16MB
  unsigned* BbitN   = (unsigned*)carve((size_t)E * (N >> 5) * 4);   // 16MB
  _Float16* Ut      = (_Float16*)carve((size_t)E * HDIM * 2);
  float* xe         = (float*)carve((size_t)E * HDIM * 4);
  float* xn         = (float*)carve((size_t)N * HDIM * 4);
  unsigned int* pooled = (unsigned int*)carve(HDIM * 4);

  // normalization + bit-pack (single f32 B pass)
  hipMemsetAsync(edge_card, 0, (size_t)E * 4, stream);
  hipMemsetAsync(beta_denom, 0, (size_t)E * 4, stream);
  hipMemsetAsync(BbitN, 0, (size_t)E * (N >> 5) * 4, stream);
  k_build<<<N, 256, 0, stream>>>(B, BbitE, BbitN, edge_card, node_deg, E, N);
  int mx = (N > E) ? N : E;
  k_pows<<<(mx + 255) / 256, 256, 0, stream>>>(edge_card, node_deg, e_alpha, v_beta, E, N);
  k_denom2<<<N, 256, 0, stream>>>(BbitE, e_alpha, v_beta, beta_denom, inv_alpha, E);
  k_invfix<<<(E + 255) / 256, 256, 0, stream>>>(beta_denom, inv_beta, E);

  // two HNHN layers
  for (int l = 0; l < 2; l++) {
    const float* X = l ? xn : x0;
    const int Kd = l ? HDIM : DIN;
    // U^T = diag(v_beta) * (X @ W01), f16 64xN
    k_sgemm<<<N / 32, 256, 32 * Kd * 4, stream>>>(X, W01[l], v_beta, Ut, N, Kd);
    // x1_pre = B^T @ U : A rows indexed by e -> BbitN (Kdim = N)
    hipMemsetAsync(xe, 0, (size_t)E * HDIM * 4, stream);
    k_wmma<<<dim3(E / 128, KSPLIT), 256, 0, stream>>>(BbitN, Ut, xe, N, N / KSPLIT);
    k_epi<<<(E * HDIM + 255) / 256, 256, 0, stream>>>(xe, inv_beta, b01[l], E);
    // V^T = diag(e_alpha) * (x1 @ W10), f16 64xE
    k_sgemm<<<E / 32, 256, 32 * HDIM * 4, stream>>>(xe, W10[l], e_alpha, Ut, E, HDIM);
    // x_pre = B @ V : A rows indexed by v -> BbitE (Kdim = E)
    hipMemsetAsync(xn, 0, (size_t)N * HDIM * 4, stream);
    k_wmma<<<dim3(N / 128, KSPLIT), 256, 0, stream>>>(BbitE, Ut, xn, E, E / KSPLIT);
    k_epi<<<(N * HDIM + 255) / 256, 256, 0, stream>>>(xn, inv_alpha, b10[l], N);
  }

  // global max pool + output
  hipMemsetAsync(pooled, 0, HDIM * 4, stream);
  k_pool<<<N / 256, 256, 0, stream>>>(xn, pooled);
  k_final<<<1, HDIM, 0, stream>>>(pooled, Wout, bout, (float*)d_out);
}
